// TopkRouter_44736379355519
// MI455X (gfx1250) — matrix operand order repaired and checked by reading, not compiled, not run on hardware
//
#include <hip/hip_runtime.h>

// ---------------------------------------------------------------------------
// MI455X (gfx1250) Top-K router.
//   Kernel 1: H = relu(X @ W1 + b1)  via v_wmma_f32_16x16x32_bf16.
//             Block tile 128x128, BK=64 -> 16 WMMAs per staging barrier,
//             wave tile 32x64 (8 accumulators) for high WMMA/LDS-byte.
//   Kernel 2: S = H @ W2 + b2 with the A-tile DMA'd by the Tensor Data Mover
//             (tensor_load_to_lds + s_wait_tensorcnt), then top-8 + softmax.
// Compute-bound (AI ~490 FLOP/B vs ~5 FLOP/B machine balance at 23.3 TB/s),
// so everything rides the WMMA pipe in bf16 with f32 accumulation.
// ---------------------------------------------------------------------------

typedef __bf16 bf16_t;
typedef __attribute__((ext_vector_type(16))) __bf16 v16bf;
typedef __attribute__((ext_vector_type(8)))  __bf16 v8bf;
typedef __attribute__((ext_vector_type(4)))  __bf16 v4bf;
typedef __attribute__((ext_vector_type(2)))  __bf16 v2bf;
typedef __attribute__((ext_vector_type(8)))  float  v8f;
typedef __attribute__((ext_vector_type(4)))  float  vf4;
typedef __attribute__((ext_vector_type(2)))  float  vf2;
typedef __attribute__((ext_vector_type(4)))  unsigned v4u;
typedef __attribute__((ext_vector_type(8)))  unsigned v8u;

#define EMBED 1024
#define NEXP  64
#define TOPK  8

// ---------------------------------------------------------------------------
// Kernel 1: X[M,1024] f32, W1[1024,1024] f32 -> H[M,1024] bf16 (relu(XW1+b1))
// 256 threads = 8 waves in a 4(M) x 2(N) grid; each wave owns 32x64
// (2 x 4 f32 16x16 accumulators). K-step 64 -> 16 WMMAs per barrier.
// ---------------------------------------------------------------------------
__global__ __launch_bounds__(256)
void router_gemm1_relu(const float* __restrict__ X, const float* __restrict__ W1,
                       const float* __restrict__ b1, bf16_t* __restrict__ H)
{
    constexpr int BM = 128, BN = 128, BK = 64;
    __shared__ __align__(64) bf16_t As[BM][BK];   // [m][k]
    __shared__ __align__(64) bf16_t Bs[BN][BK];   // [n][k] (transposed W1 tile)

    const int tid   = threadIdx.x;
    const int lane  = tid & 31;          // wave32
    const int wave  = tid >> 5;
    const int m_blk = blockIdx.x * BM;
    const int n_blk = blockIdx.y * BN;
    const int wm    = (wave & 3) * 32;   // wave M offset in block tile
    const int wn    = (wave >> 2) * 64;  // wave N offset
    const int halfK = lane >> 4;         // 0: lanes 0-15, 1: lanes 16-31
    const int l15   = lane & 15;

    v8f acc[2][4] = {};

    for (int k0 = 0; k0 < EMBED; k0 += BK) {
        // Stage A: 128x64 f32 -> bf16 via packed cvt. 2048 float4 / 256 thr.
        #pragma unroll
        for (int i = 0; i < 8; ++i) {
            const int idx = tid + i * 256;         // 0..2047 float4 chunks
            const int r   = idx >> 4;              // row 0..127
            const int kq  = (idx & 15) << 2;       // k 0,4,...,60
            const vf4 v = *(const vf4*)(X + (size_t)(m_blk + r) * EMBED + k0 + kq);
            *(v4bf*)&As[r][kq] = __builtin_convertvector(v, v4bf);
        }
        // Stage B transposed: pair two k rows so each LDS store is one b32.
        // 128 n x 64 k = 4096 b32 stores / 256 threads.
        #pragma unroll
        for (int i = 0; i < 16; ++i) {
            const int idx = tid + i * 256;         // 0..4095
            const int kp  = (idx >> 7) << 1;       // k 0,2,...,62
            const int n   = idx & 127;
            vf2 f;
            f.x = W1[(size_t)(k0 + kp)     * EMBED + n_blk + n];
            f.y = W1[(size_t)(k0 + kp + 1) * EMBED + n_blk + n];
            *(v2bf*)&Bs[n][kp] = __builtin_convertvector(f, v2bf);
        }
        __syncthreads();

        // Prefetch next A tile into cache (gfx1250 global_prefetch_b8).
        if (k0 + BK < EMBED) {
            __builtin_prefetch(X + (size_t)(m_blk + (tid >> 1)) * EMBED + k0 + BK
                                 + ((tid & 1) << 5), 0, 0);
        }

        #pragma unroll
        for (int ks = 0; ks < BK; ks += 32) {
            v16bf afrag[2], bfrag[4];
            #pragma unroll
            for (int ai = 0; ai < 2; ++ai) {
                const int m = wm + ai * 16 + l15;
                afrag[ai] = *(const v16bf*)&As[m][ks + halfK * 16];
            }
            #pragma unroll
            for (int bi = 0; bi < 4; ++bi) {
                const int n = wn + bi * 16 + l15;
                bfrag[bi] = *(const v16bf*)&Bs[n][ks + halfK * 16];
            }
            #pragma unroll
            for (int ai = 0; ai < 2; ++ai)
                #pragma unroll
                for (int bi = 0; bi < 4; ++bi)
                    acc[ai][bi] = __builtin_amdgcn_wmma_f32_16x16x32_bf16(
                        false, afrag[ai], false, bfrag[bi], (short)0,
                        acc[ai][bi], false, false);
        }
        __syncthreads();
    }

    // Epilogue: bias + relu, store bf16 H.
    #pragma unroll
    for (int ai = 0; ai < 2; ++ai) {
        #pragma unroll
        for (int bi = 0; bi < 4; ++bi) {
            const int n    = n_blk + wn + bi * 16 + l15;
            const float bv = b1[n];
            #pragma unroll
            for (int e = 0; e < 8; ++e) {
                const int m = m_blk + wm + ai * 16 + e + halfK * 8;
                float v = acc[ai][bi][e] + bv;
                v = v > 0.0f ? v : 0.0f;
                H[(size_t)m * EMBED + n] = (bf16_t)v;
            }
        }
    }
}

// ---------------------------------------------------------------------------
// Kernel 2: scores = H @ W2 + b2 (M-tile 128, N=64). The 128x32 bf16 A tile
// is fetched each K-step by the Tensor Data Mover (one DMA descriptor issued
// by wave 0) instead of per-thread staging. Then per-token top-8 + softmax.
// ---------------------------------------------------------------------------
__global__ __launch_bounds__(256)
void router_gemm2_topk(const bf16_t* __restrict__ H, const float* __restrict__ W2,
                       const float* __restrict__ b2, float* __restrict__ probs,
                       float* __restrict__ idx_out, int Mtot)
{
    constexpr int BM = 128, BK = 32;
    __shared__ __align__(64) bf16_t As[BM][BK];       // TDM destination
    __shared__ __align__(64) bf16_t Bs[NEXP][BK];     // [n][k]
    __shared__ __align__(64) float  Sc[BM][NEXP + 1]; // scores, +1 pad

    const int tid   = threadIdx.x;
    const int lane  = tid & 31;
    const int wave  = tid >> 5;
    const int m_blk = blockIdx.x * BM;
    const int wm    = wave * 16;
    const int halfK = lane >> 4;
    const int l15   = lane & 15;

    // --- Loop-invariant TDM descriptor group 1 (CDNA5 ISA)
    // data_size=2B, tensor_dim0=EMBED, tensor_dim1=Mtot, tile_dim0=BK,
    // tile_dim1=BM, tensor_dim0_stride=EMBED. No padding / iterate / barrier.
    v8u g1;
    g1[0] = 0x00010000u;                               // data_size=1 (2 bytes)
    g1[1] = ((unsigned)EMBED & 0xFFFFu) << 16;         // tensor_dim0 lo16
    g1[2] = (((unsigned)EMBED >> 16) & 0xFFFFu)        // tensor_dim0 hi16
          | (((unsigned)Mtot & 0xFFFFu) << 16);        // tensor_dim1 lo16
    g1[3] = (((unsigned)Mtot >> 16) & 0xFFFFu)         // tensor_dim1 hi16
          | ((unsigned)BK << 16);                      // tile_dim0
    g1[4] = (unsigned)BM;                              // tile_dim1 (tile_dim2=0)
    g1[5] = (unsigned)EMBED;                           // tensor_dim0_stride lo32
    g1[6] = 0u;                                        // stride hi16, dim1_stride lo16
    g1[7] = 0u;

    const unsigned long long gbase =
        (unsigned long long)(size_t)(H + (size_t)m_blk * EMBED);
    const unsigned ldsA = (unsigned)(size_t)&As[0][0];

    v8f acc[4] = {};

    for (int k0 = 0; k0 < EMBED; k0 += BK) {
        // Wave 0 issues the TDM load for the A tile (EXEC-independent DMA).
        if (wave == 0) {
            const unsigned long long ga = gbase + (unsigned long long)k0 * 2ull;
            v4u g0;
            g0[0] = 1u;                                          // count=1
            g0[1] = ldsA;                                        // lds_addr
            g0[2] = (unsigned)ga;                                // global_addr lo
            g0[3] = ((unsigned)(ga >> 32) & 0x01FFFFFFu)         // global_addr hi
                  | (2u << 30);                                  // type=2 (image)
            asm volatile("tensor_load_to_lds %0, %1"
                         :: "s"(g0), "s"(g1) : "memory");
        }
        // All threads stage B: W2[k0..+31][0..63] f32 -> Bs[n][k] bf16,
        // packed cvt + b32 transposed stores.
        #pragma unroll
        for (int i = 0; i < 4; ++i) {
            const int idx = tid + i * 256;       // 0..1023
            const int kp  = (idx >> 6) << 1;     // k 0,2,...,30
            const int n   = idx & 63;
            vf2 f;
            f.x = W2[(size_t)(k0 + kp)     * NEXP + n];
            f.y = W2[(size_t)(k0 + kp + 1) * NEXP + n];
            *(v2bf*)&Bs[n][kp] = __builtin_convertvector(f, v2bf);
        }
        if (wave == 0) __builtin_amdgcn_s_wait_tensorcnt(0);
        __syncthreads();

        v16bf a;
        {
            const int m = wm + l15;
            a = *(const v16bf*)&As[m][halfK * 16];
        }
        #pragma unroll
        for (int bi = 0; bi < 4; ++bi) {
            const int n = bi * 16 + l15;
            v16bf b = *(const v16bf*)&Bs[n][halfK * 16];
            acc[bi] = __builtin_amdgcn_wmma_f32_16x16x32_bf16(
                false, a, false, b, (short)0, acc[bi], false, false);
        }
        __syncthreads();
    }

    // Scores (+bias) to LDS.
    #pragma unroll
    for (int bi = 0; bi < 4; ++bi) {
        const int n    = bi * 16 + l15;
        const float bv = b2[n];
        #pragma unroll
        for (int e = 0; e < 8; ++e) {
            const int mt = wm + e + halfK * 8;   // token within block tile
            Sc[mt][n] = acc[bi][e] + bv;
        }
    }
    __syncthreads();

    // Top-8 + softmax: one token per thread (first 128 threads).
    if (tid < BM) {
        const int token = m_blk + tid;
        float* row = &Sc[tid][0];

        float vals[TOPK];
        int   ids[TOPK];
        #pragma unroll
        for (int j = 0; j < TOPK; ++j) {
            float best = -__builtin_inff();
            int   bidx = 0;
            for (int n = 0; n < NEXP; ++n) {
                const float v = row[n];
                if (v > best) { best = v; bidx = n; }   // first-max = JAX tie-break
            }
            vals[j] = best;
            ids[j]  = bidx;
            row[bidx] = -__builtin_inff();
        }

        const float mx = vals[0];
        float ex[TOPK];
        float sum = 0.0f;
        #pragma unroll
        for (int j = 0; j < TOPK; ++j) { ex[j] = expf(vals[j] - mx); sum += ex[j]; }
        const float inv = 1.0f / sum;

        float* dst = probs + (size_t)token * NEXP;
        #pragma unroll
        for (int q = 0; q < NEXP / 4; ++q)
            ((float4*)dst)[q] = make_float4(0.f, 0.f, 0.f, 0.f);
        #pragma unroll
        for (int j = 0; j < TOPK; ++j)
            dst[ids[j]] = ex[j] * inv;          // masked softmax: others exactly 0

        #pragma unroll
        for (int j = 0; j < TOPK; ++j)
            idx_out[(size_t)token * TOPK + j] = (float)ids[j];
    }
}

// ---------------------------------------------------------------------------
extern "C" void kernel_launch(void* const* d_in, const int* in_sizes, int n_in,
                              void* d_out, int out_size, void* d_ws, size_t ws_size,
                              hipStream_t stream) {
    const float* X  = (const float*)d_in[0];   // [8,4096,1024]
    const float* W1 = (const float*)d_in[1];   // [1024,1024]
    const float* b1 = (const float*)d_in[2];   // [1024]
    const float* W2 = (const float*)d_in[3];   // [1024,64]
    const float* b2 = (const float*)d_in[4];   // [64]

    const int M = in_sizes[0] / EMBED;         // 32768 tokens

    float* probs   = (float*)d_out;                         // [M,64]
    float* idx_out = (float*)d_out + (size_t)M * NEXP;      // [M,8] as floats

    bf16_t* H = (bf16_t*)d_ws;                 // [M,1024] bf16 (64 MB)

    dim3 blk(256);
    dim3 g1(M / 128, EMBED / 128);
    router_gemm1_relu<<<g1, blk, 0, stream>>>(X, W1, b1, H);

    dim3 g2(M / 128);
    router_gemm2_topk<<<g2, blk, 0, stream>>>(H, W2, b2, probs, idx_out, M);
}
